// MambaBlock_40922448396738
// MI455X (gfx1250) — matrix operand order repaired and checked by reading, not compiled, not run on hardware
//
#include <hip/hip_runtime.h>
#include <hip/hip_bf16.h>
#include <math.h>

// ---------------------------------------------------------------------------
// Mamba block forward for MI455X (gfx1250, wave32, WMMA, double-buffered
// async->LDS staging). D_MODEL=1024, D_STATE=16, D_CONV=4, B=2, L=2048.
// ---------------------------------------------------------------------------

#define DM   1024      // d_model
#define DS   16        // d_state
#define DC   4         // d_conv
#define BB   2         // batch
#define LL   2048      // seq len
#define ROWS (BB * LL) // 4096 GEMM rows

typedef __bf16 bf16_t;
typedef __attribute__((ext_vector_type(16))) __bf16 v16bf;
typedef __attribute__((ext_vector_type(8)))  __bf16 v8bf;
typedef __attribute__((ext_vector_type(8)))  float  v8f;

__device__ __forceinline__ bf16_t f2bf(float f) { return (bf16_t)f; }

// 16-byte global -> LDS direct copy (ASYNCcnt-tracked, no VGPR round trip).
__device__ __forceinline__ void async_copy_b128(unsigned int lds_off,
                                                unsigned long long gaddr) {
  asm volatile("global_load_async_to_lds_b128 %0, %1, off"
               :: "v"(lds_off), "v"(gaddr) : "memory");
}
__device__ __forceinline__ void wait_async_le2() {
  asm volatile("s_wait_asynccnt 2" ::: "memory");
}
__device__ __forceinline__ void wait_async0() {
  asm volatile("s_wait_asynccnt 0" ::: "memory");
}

// ---------------------------------------------------------------------------
// f32 -> bf16 bulk convert, 8 elements / thread (n % 8 == 0 for all uses)
// ---------------------------------------------------------------------------
__global__ void cvt_f32_bf16_kernel(const float* __restrict__ in,
                                    bf16_t* __restrict__ out, int n) {
  int i = blockIdx.x * blockDim.x + threadIdx.x;
  if (i * 8 < n) {
    float4 a = ((const float4*)in)[i * 2];
    float4 b = ((const float4*)in)[i * 2 + 1];
    v8bf o;
    o[0] = f2bf(a.x); o[1] = f2bf(a.y); o[2] = f2bf(a.z); o[3] = f2bf(a.w);
    o[4] = f2bf(b.x); o[5] = f2bf(b.y); o[6] = f2bf(b.z); o[7] = f2bf(b.w);
    *(v8bf*)(out + i * 8) = o;
  }
}

// ---------------------------------------------------------------------------
// bf16 GEMM, f32 accumulate, + bias:  C[M,N] = A[M,K] @ B[K,N] + b
//
// Block = 256 threads (8 waves). Block tile 128x128; wave tile 32x64 =
// 2x4 v_wmma_f32_16x16x32_bf16. Requirements: M % 128 == 0, K % 32 == 0,
// N % 8 == 0. FULL_N=true when N % 128 == 0 (drops the edge guard).
//
// Double-buffered pipeline: A tiles stream memory->LDS via
// global_load_async_to_lds_b128 concurrently with WMMA on the other buffer
// (s_wait_asynccnt 2 = only the current tile's two per-wave async ops must
// have retired; async completion is in-order). B tiles are loaded to
// registers early and scatter-stored transposed (BsT[n][k]) after the math,
// so global latency also hides under the WMMAs. K-stride padded to 40
// halfwords (80B) -> the 16 row-strided ds_load_b128 lanes of a fragment
// hit disjoint bank quads (20*r mod 64 distinct): conflict-free.
// ---------------------------------------------------------------------------
#define TMB 128
#define TNB 128
#define TKB 32
#define KPAD 40

template <bool FULL_N>
__global__ __launch_bounds__(256)
void gemm_bias_bf16_wmma(const bf16_t* __restrict__ A,
                         const bf16_t* __restrict__ Bw,
                         const float*  __restrict__ bias,
                         float*        __restrict__ C,
                         int M, int N, int K) {
  __shared__ __align__(16) bf16_t As [2][TMB][KPAD];   // 20 KB
  __shared__ __align__(16) bf16_t BsT[2][TNB][KPAD];   // 20 KB

  const int tid  = threadIdx.x;
  const int wave = tid >> 5;
  const int lane = tid & 31;
  const int wm   = wave >> 1;      // wave row (0..3) -> 32 rows
  const int wn   = wave & 1;       // wave col (0..1) -> 64 cols
  const int mlan = lane & 15;
  const int lhi  = lane >> 4;

  const int rowBase = blockIdx.y * TMB;
  const int colBase = blockIdx.x * TNB;

  const unsigned int asBase    = (unsigned int)(size_t)&As[0][0][0];
  const unsigned int abufBytes = TMB * KPAD * 2u;

  auto issueA = [&](int k0, int buf) {
    #pragma unroll
    for (int rep = 0; rep < 2; ++rep) {
      int j  = tid + rep * 256;              // chunk 0..511
      int r  = j >> 2;                       // row in tile
      int c8 = (j & 3) << 3;                 // k offset
      unsigned long long g =
          (unsigned long long)(size_t)(A + (size_t)(rowBase + r) * K + k0 + c8);
      async_copy_b128(asBase + (unsigned int)buf * abufBytes +
                      (unsigned int)(r * KPAD + c8) * 2u, g);
    }
  };
  auto loadB = [&](int k0, v8bf vb[2]) {
    #pragma unroll
    for (int rep = 0; rep < 2; ++rep) {
      int j  = tid + rep * 256;
      int k  = j >> 4;                       // k row (0..31)
      int c8 = (j & 15) << 3;                // col offset
      int gc = colBase + c8;
      if (FULL_N || gc + 8 <= N)
        vb[rep] = *(const v8bf*)(Bw + (size_t)(k0 + k) * N + gc);
      else
        vb[rep] = (v8bf){};                  // zero-fill ragged N (N=48)
    }
  };
  auto storeB = [&](int buf, const v8bf vb[2]) {
    #pragma unroll
    for (int rep = 0; rep < 2; ++rep) {
      int j  = tid + rep * 256;
      int k  = j >> 4;
      int c8 = (j & 15) << 3;
      #pragma unroll
      for (int i = 0; i < 8; ++i) BsT[buf][c8 + i][k] = vb[rep][i];
    }
  };

  v8f acc[2][4];
  #pragma unroll
  for (int mt = 0; mt < 2; ++mt)
    #pragma unroll
    for (int nt = 0; nt < 4; ++nt)
      acc[mt][nt] = (v8f){};

  // ---- prologue: stage tile 0 into buffer 0 ------------------------------
  v8bf vb[2];
  issueA(0, 0);
  loadB(0, vb);
  storeB(0, vb);

  const int nk = K / TKB;
  for (int kt = 0; kt < nk; ++kt) {
    const int cur = kt & 1;
    const bool hasNext = (kt + 1 < nk);
    if (hasNext) {
      issueA((kt + 1) * TKB, cur ^ 1);       // DMA overlaps this tile's math
      loadB((kt + 1) * TKB, vb);             // global latency hides too
      wait_async_le2();                      // current tile's A is resident
    } else {
      wait_async0();
    }
    __syncthreads();

    // ---- fragments: 2 aligned ds_load_b128 each --------------------------
    // A lane layout: m=lane&15, elem e -> K=(e&7)+(e>>3)*16+(lane>>4)*8
    // B lane layout: n=lane&15, elem e -> K=(e&7)+(e>>3)*16+(lane>>4)*8
    v16bf afrag[2], bfrag[4];
    #pragma unroll
    for (int mt = 0; mt < 2; ++mt) {
      int ar = wm * 32 + mt * 16 + mlan;
      v8bf lo = *(const v8bf*)&As[cur][ar][lhi * 8];
      v8bf hi = *(const v8bf*)&As[cur][ar][16 + lhi * 8];
      afrag[mt] = __builtin_shufflevector(lo, hi, 0,1,2,3,4,5,6,7,
                                                  8,9,10,11,12,13,14,15);
    }
    #pragma unroll
    for (int nt = 0; nt < 4; ++nt) {
      int bc = wn * 64 + nt * 16 + mlan;
      v8bf lo = *(const v8bf*)&BsT[cur][bc][lhi * 8];
      v8bf hi = *(const v8bf*)&BsT[cur][bc][16 + lhi * 8];
      bfrag[nt] = __builtin_shufflevector(lo, hi, 0,1,2,3,4,5,6,7,
                                                  8,9,10,11,12,13,14,15);
    }

    // ---- 8 x v_wmma_f32_16x16x32_bf16 per wave per K-step ----------------
    #pragma unroll
    for (int mt = 0; mt < 2; ++mt)
      #pragma unroll
      for (int nt = 0; nt < 4; ++nt)
        acc[mt][nt] = __builtin_amdgcn_wmma_f32_16x16x32_bf16(
            false, afrag[mt], false, bfrag[nt],
            (short)0, acc[mt][nt], false, false);

    if (hasNext) storeB(cur ^ 1, vb);        // transpose-scatter after math
    __syncthreads();
  }

  // ---- epilogue: C/D layout m = r + 8*(lane>>4), n = lane&15 -------------
  #pragma unroll
  for (int mt = 0; mt < 2; ++mt) {
    #pragma unroll
    for (int nt = 0; nt < 4; ++nt) {
      int col = colBase + wn * 64 + nt * 16 + mlan;
      if (col < N) {
        float bv = bias[col];
        #pragma unroll
        for (int r = 0; r < 8; ++r) {
          int row = rowBase + wm * 32 + mt * 16 + r + (lhi << 3);
          C[(size_t)row * N + col] = acc[mt][nt][r] + bv;
        }
      }
    }
  }
}

// ---------------------------------------------------------------------------
// Causal depthwise conv1d (k=4) + bias + SiLU; reads the u-half of xz.
// Writes u in f32 (scan) and bf16 (W_xp GEMM).
// ---------------------------------------------------------------------------
__global__ void conv_silu_kernel(const float* __restrict__ xz,
                                 const float* __restrict__ Wc,
                                 const float* __restrict__ bc,
                                 float*  __restrict__ u32,
                                 bf16_t* __restrict__ ubf) {
  int idx = blockIdx.x * blockDim.x + threadIdx.x;   // over B*L*d
  if (idx >= ROWS * DM) return;
  int d = idx % DM;
  int t = (idx / DM) % LL;
  int b = idx / (DM * LL);

  float acc = bc[d];
  #pragma unroll
  for (int j = 0; j < DC; ++j) {
    int tt = t - (DC - 1) + j;
    if (tt >= 0)
      acc += Wc[d * DC + j] * xz[((size_t)(b * LL + tt)) * (2 * DM) + d];
  }
  float s = acc / (1.0f + __expf(-acc));   // silu
  u32[idx] = s;
  ubf[idx] = f2bf(s);
}

// ---------------------------------------------------------------------------
// dt = softplus(dt_in @ W_dt + b_dt).  K=16 -> VALU dot.
// ---------------------------------------------------------------------------
__global__ void dt_softplus_kernel(const float* __restrict__ xdb,
                                   const float* __restrict__ Wdt,
                                   const float* __restrict__ bdt,
                                   float* __restrict__ dt) {
  int d   = blockIdx.x * blockDim.x + threadIdx.x;
  int row = blockIdx.y;
  if (d >= DM) return;
  float acc = bdt[d];
  #pragma unroll
  for (int i = 0; i < DS; ++i)
    acc += xdb[(size_t)row * (3 * DS) + i] * Wdt[i * DM + d];
  float sp = (acc > 20.0f) ? acc : log1pf(__expf(acc));
  dt[(size_t)row * DM + d] = sp;
}

// ---------------------------------------------------------------------------
// Fused selective scan + output gate. One thread per (b,d); 16 SSM states
// in registers; dA/dBu never materialized. Emits y*silu(z) as bf16.
// ---------------------------------------------------------------------------
__global__ __launch_bounds__(256)
void scan_gate_kernel(const float* __restrict__ dt,
                      const float* __restrict__ u,
                      const float* __restrict__ xdb,
                      const float* __restrict__ A_log,
                      const float* __restrict__ xz,
                      bf16_t* __restrict__ ygbf) {
  int d = blockIdx.x * blockDim.x + threadIdx.x;
  int b = blockIdx.y;
  if (d >= DM) return;

  float Ac[DS], h[DS];
  #pragma unroll
  for (int n = 0; n < DS; ++n) {
    Ac[n] = -__expf(A_log[d * DS + n]);
    h[n]  = 0.0f;
  }

  for (int t = 0; t < LL; ++t) {
    size_t row = (size_t)b * LL + t;
    if (t + 1 < LL) {   // pull next timestep toward WGP (global_prefetch_b8)
      __builtin_prefetch(&dt[(row + 1) * DM + d], 0, 3);
      __builtin_prefetch(&u[(row + 1) * DM + d], 0, 3);
    }
    float dtv = dt[row * DM + d];
    float uv  = u[row * DM + d];
    float dtu = dtv * uv;
    const float* xr = xdb + row * (3 * DS);   // [dt_in | B | C], block-uniform
    float y = 0.0f;
    #pragma unroll
    for (int n = 0; n < DS; ++n) {
      float e = __expf(dtv * Ac[n]);
      h[n] = e * h[n] + dtu * xr[DS + n];     // B
      y += h[n] * xr[2 * DS + n];             // C
    }
    float zv = xz[row * (2 * DM) + DM + d];
    float g  = zv / (1.0f + __expf(-zv));     // silu(z)
    ygbf[row * DM + d] = f2bf(y * g);
  }
}

// ---------------------------------------------------------------------------
// Launcher
// ---------------------------------------------------------------------------
extern "C" void kernel_launch(void* const* d_in, const int* in_sizes, int n_in,
                              void* d_out, int out_size, void* d_ws, size_t ws_size,
                              hipStream_t stream) {
  const float* x      = (const float*)d_in[0];   // (B,L,d)
  const float* W_in   = (const float*)d_in[1];   // (d, 2d)
  const float* b_in   = (const float*)d_in[2];   // (2d)
  const float* W_conv = (const float*)d_in[3];   // (d,1,k)
  const float* b_conv = (const float*)d_in[4];   // (d)
  const float* W_xp   = (const float*)d_in[5];   // (d, 3n)
  const float* b_xp   = (const float*)d_in[6];   // (3n)
  const float* W_dt   = (const float*)d_in[7];   // (n, d)
  const float* b_dt   = (const float*)d_in[8];   // (d)
  const float* A_log  = (const float*)d_in[9];   // (d, n)
  const float* W_out  = (const float*)d_in[10];  // (d, d)
  const float* b_out  = (const float*)d_in[11];  // (d)
  float* out = (float*)d_out;                    // (B,L,d)

  // ---- workspace carve-out (256B aligned) ----
  size_t off = 0;
  auto carve = [&](size_t bytes) -> void* {
    void* p = (void*)((char*)d_ws + off);
    off += (bytes + 255) & ~(size_t)255;
    return p;
  };
  bf16_t* x_bf    = (bf16_t*)carve((size_t)ROWS * DM * sizeof(bf16_t));
  bf16_t* Win_bf  = (bf16_t*)carve((size_t)DM * 2 * DM * sizeof(bf16_t));
  bf16_t* Wxp_bf  = (bf16_t*)carve((size_t)DM * 3 * DS * sizeof(bf16_t));
  bf16_t* Wout_bf = (bf16_t*)carve((size_t)DM * DM * sizeof(bf16_t));
  float*  xz      = (float*)carve((size_t)ROWS * 2 * DM * sizeof(float));
  float*  u32     = (float*)carve((size_t)ROWS * DM * sizeof(float));
  bf16_t* ubf     = (bf16_t*)carve((size_t)ROWS * DM * sizeof(bf16_t));
  float*  xdb     = (float*)carve((size_t)ROWS * 3 * DS * sizeof(float));
  float*  dt      = (float*)carve((size_t)ROWS * DM * sizeof(float));
  bf16_t* ygbf    = (bf16_t*)carve((size_t)ROWS * DM * sizeof(bf16_t));
  (void)ws_size;

  // ---- 1. downconvert GEMM operands to bf16 (8 elems/thread) ----
  {
    int n;
    n = ROWS * DM;
    cvt_f32_bf16_kernel<<<(n / 8 + 255) / 256, 256, 0, stream>>>(x, x_bf, n);
    n = DM * 2 * DM;
    cvt_f32_bf16_kernel<<<(n / 8 + 255) / 256, 256, 0, stream>>>(W_in, Win_bf, n);
    n = DM * 3 * DS;
    cvt_f32_bf16_kernel<<<(n / 8 + 255) / 256, 256, 0, stream>>>(W_xp, Wxp_bf, n);
    n = DM * DM;
    cvt_f32_bf16_kernel<<<(n / 8 + 255) / 256, 256, 0, stream>>>(W_out, Wout_bf, n);
  }

  // ---- 2. xz = x @ W_in + b_in   (4096 x 2048 x K1024) ----
  {
    dim3 grid(2 * DM / TNB, ROWS / TMB);
    gemm_bias_bf16_wmma<true><<<grid, 256, 0, stream>>>(x_bf, Win_bf, b_in, xz,
                                                        ROWS, 2 * DM, DM);
  }

  // ---- 3. causal depthwise conv + SiLU ----
  {
    int n = ROWS * DM;
    conv_silu_kernel<<<(n + 255) / 256, 256, 0, stream>>>(xz, W_conv, b_conv,
                                                          u32, ubf);
  }

  // ---- 4. x_db = u @ W_xp + b_xp   (4096 x 48 x K1024) ----
  {
    dim3 grid(1, ROWS / TMB);
    gemm_bias_bf16_wmma<false><<<grid, 256, 0, stream>>>(ubf, Wxp_bf, b_xp, xdb,
                                                         ROWS, 3 * DS, DM);
  }

  // ---- 5. dt = softplus(dt_in @ W_dt + b_dt) ----
  {
    dim3 grid(DM / 256, ROWS);
    dt_softplus_kernel<<<grid, 256, 0, stream>>>(xdb, W_dt, b_dt, dt);
  }

  // ---- 6. fused selective scan + gate ----
  {
    dim3 grid(DM / 256, BB);
    scan_gate_kernel<<<grid, 256, 0, stream>>>(dt, u32, xdb, A_log, xz, ygbf);
  }

  // ---- 7. out = (y * silu(z)) @ W_out + b_out   (4096 x 1024 x K1024) ----
  {
    dim3 grid(DM / TNB, ROWS / TMB);
    gemm_bias_bf16_wmma<true><<<grid, 256, 0, stream>>>(ygbf, Wout_bf, b_out, out,
                                                        ROWS, DM, DM);
  }
}
